// CapsuleNetwork_11982958756250
// MI455X (gfx1250) — matrix operand order repaired and checked by reading, not compiled, not run on hardware
//
#include <hip/hip_runtime.h>
#include <hip/hip_bf16.h>
#include <stdint.h>
#include <stddef.h>

typedef __bf16 bf16_t;
typedef __attribute__((ext_vector_type(4)))  __bf16 v4bf;
typedef __attribute__((ext_vector_type(8)))  __bf16 v8bf;
typedef __attribute__((ext_vector_type(16))) __bf16 v16bf;
typedef __attribute__((ext_vector_type(8)))  float  v8f;

#define B_SZ    1024
#define IN_DIM  768
#define PROJ    9216
#define NCLS    5
#define NCAPS   512
#define PDIM    8
#define RDIM    16
#define CONV_K  2304          // 256 in-ch * 9 taps
#define CONV_N  256           // out channels (32 caps * 8)
#define CONV_M  (B_SZ * 16)   // batch * 4x4 output positions

static __device__ __forceinline__ v16bf mk16(v8bf lo, v8bf hi) {
    union { v16bf v; v8bf h[2]; } u;
    u.h[0] = lo; u.h[1] = hi;
    return u.v;
}

// ---------------- elementwise f32 -> bf16 (x4 vectorized) ----------------
__global__ void cvt4_f32_bf16(const float* __restrict__ in,
                              bf16_t* __restrict__ out, int n4) {
    int i = blockIdx.x * blockDim.x + threadIdx.x;
    if (i < n4) {
        float4 f = ((const float4*)in)[i];
        v4bf o = { (__bf16)f.x, (__bf16)f.y, (__bf16)f.z, (__bf16)f.w };
        ((v4bf*)out)[i] = o;
    }
}

// ---------------- tiled transpose + convert: in[R][C] f32 -> out[C][R] bf16 ----------------
__global__ __launch_bounds__(256) void transpose_cvt(const float* __restrict__ in,
                                                     bf16_t* __restrict__ out,
                                                     int R, int C) {
    __shared__ bf16_t tile[32][33];
    const int tx = threadIdx.x & 31, ty = threadIdx.x >> 5;  // 32 x 8
    const int c0 = blockIdx.x * 32, r0 = blockIdx.y * 32;
#pragma unroll
    for (int j = 0; j < 4; ++j) {
        int r = ty + j * 8;
        tile[r][tx] = (bf16_t)in[(size_t)(r0 + r) * C + (c0 + tx)];
    }
    __syncthreads();
#pragma unroll
    for (int j = 0; j < 4; ++j) {
        int r = ty + j * 8;
        out[(size_t)(c0 + r) * R + (r0 + tx)] = tile[tx][r];
    }
}

// ---------------- bf16 WMMA GEMM ----------------
// C[M][N] = A[M][K] * BT[N][K]^T (+bias, +relu); A optionally gathered as conv patches
// block: 256 threads = 8 waves, tile 128(M) x 128(N), K-step 32
// wave (wm in 0..3, wn in 0..1) owns rows wm*32..+32, cols wn*64..+64 => 2x4 WMMA tiles
template <bool RELU, bool OUT_BF16, bool CONV_GATHER>
__global__ __launch_bounds__(256) void gemm_wmma(
    const bf16_t* __restrict__ A,     // M x K row-major (unused if CONV_GATHER)
    const bf16_t* __restrict__ Hsrc,  // [B][256*36] conv input (if CONV_GATHER)
    const bf16_t* __restrict__ BT,    // N x K row-major
    const float*  __restrict__ bias,  // len N or nullptr
    float*  __restrict__ Cf,
    bf16_t* __restrict__ Cb,
    int M, int N, int K)
{
    __shared__ __align__(16) bf16_t sA[2][128][40];   // 40-elem (80B) padded rows
    __shared__ __align__(16) bf16_t sB[2][128][40];

    const int t    = threadIdx.x;
    const int lane = t & 31;
    const int wid  = t >> 5;
    const int l16  = lane & 15;
    const int hi   = lane >> 4;
    const int wm   = wid >> 1;   // 0..3
    const int wn   = wid & 1;    // 0..1
    const int row0 = blockIdx.y * 128;
    const int col0 = blockIdx.x * 128;
    const int KT   = K >> 5;

    v8f acc[2][4];
#pragma unroll
    for (int a = 0; a < 2; ++a)
#pragma unroll
        for (int b = 0; b < 4; ++b)
#pragma unroll
            for (int e = 0; e < 8; ++e) acc[a][b][e] = 0.f;

    v8bf ra[2], rb[2];   // register staging for double-buffered LDS

    auto loadTiles = [&](int kt) {
        const int kbase = kt << 5;
#pragma unroll
        for (int i = 0; i < 2; ++i) {
            int idx = (i * 256 + t) * 8;     // 0..4095 of 128x32 tile
            int r = idx >> 5, c = idx & 31;
            if (CONV_GATHER) {
                // implicit im2col: m -> (batch, 4x4 pos), k -> (in-ch, 3x3 tap)
                int m  = row0 + r;
                int bb = m >> 4, p = m & 15, ph = p >> 2, pw = p & 3;
                union { v8bf v; bf16_t e[8]; } tmp;
#pragma unroll
                for (int e = 0; e < 8; ++e) {
                    int k  = kbase + c + e;
                    int ic = k / 9; int rr = k - ic * 9;
                    int kh = rr / 3; int kw = rr - kh * 3;
                    tmp.e[e] = Hsrc[(size_t)bb * 9216 + ic * 36 + (ph + kh) * 6 + (pw + kw)];
                }
                ra[i] = tmp.v;
            } else {
                ra[i] = *(const v8bf*)(A + (size_t)(row0 + r) * K + kbase + c);
            }
            rb[i] = *(const v8bf*)(BT + (size_t)(col0 + r) * K + kbase + c);
        }
    };

    auto storeTiles = [&](int buf) {
#pragma unroll
        for (int i = 0; i < 2; ++i) {
            int idx = (i * 256 + t) * 8;
            int r = idx >> 5, c = idx & 31;
            *(v8bf*)&sA[buf][r][c] = ra[i];
            *(v8bf*)&sB[buf][r][c] = rb[i];
        }
    };

    auto compute = [&](int buf) {
        v16bf af[2], bfm[4];
        // A fragment 16x32: lanes 0-15 row M, elems 0-7=K(hi*8..), 8-15=K(16+hi*8..)
#pragma unroll
        for (int mt = 0; mt < 2; ++mt) {
            const bf16_t* p = &sA[buf][wm * 32 + mt * 16 + l16][0];
            af[mt] = mk16(*(const v8bf*)(p + hi * 8), *(const v8bf*)(p + 16 + hi * 8));
        }
        // B fragment 32x16: lane holds col N=l16, elems 0-7=K(hi*8..), 8-15=K(16+hi*8..)
#pragma unroll
        for (int nt = 0; nt < 4; ++nt) {
            const bf16_t* p = &sB[buf][wn * 64 + nt * 16 + l16][hi * 8];
            bfm[nt] = mk16(*(const v8bf*)p, *(const v8bf*)(p + 16));
        }
#pragma unroll
        for (int mt = 0; mt < 2; ++mt)
#pragma unroll
            for (int nt = 0; nt < 4; ++nt)
                acc[mt][nt] = __builtin_amdgcn_wmma_f32_16x16x32_bf16(
                    false, af[mt], false, bfm[nt], (short)0, acc[mt][nt], false, false);
    };

    loadTiles(0);
    storeTiles(0);
    __syncthreads();
    for (int kt = 0; kt < KT; ++kt) {
        const int cur = kt & 1;
        if (kt + 1 < KT) loadTiles(kt + 1);   // global loads overlap WMMA
        compute(cur);
        if (kt + 1 < KT) {
            __syncthreads();
            storeTiles(cur ^ 1);
            __syncthreads();
        }
    }

    // epilogue: C/D layout — elem r of v8f at lane: row = 8*hi + r, col = l16
#pragma unroll
    for (int mt = 0; mt < 2; ++mt) {
#pragma unroll
        for (int nt = 0; nt < 4; ++nt) {
            int gcol  = col0 + wn * 64 + nt * 16 + l16;
            int growb = row0 + wm * 32 + mt * 16 + hi * 8;
            float bv = bias ? bias[gcol] : 0.f;
#pragma unroll
            for (int r = 0; r < 8; ++r) {
                float v = acc[mt][nt][r] + bv;
                if (RELU) v = fmaxf(v, 0.f);
                if (OUT_BF16) Cb[(size_t)(growb + r) * N + gcol] = (bf16_t)v;
                else          Cf[(size_t)(growb + r) * N + gcol] = v;
            }
        }
    }
}

// ---------------- fused squash + u_hat + dynamic routing ----------------
// one workgroup per batch item; u_hat (5x16x512 f32 = 160KB) lives in LDS
__global__ __launch_bounds__(256) void routing_kernel(
    const float* __restrict__ G,    // [B*16][256] conv-GEMM output (bias applied)
    const float* __restrict__ Wr,   // [512][5][16][8]
    float* __restrict__ out_probs,  // [B][5]
    float* __restrict__ out_v)      // [B][5][16]
{
    __shared__ float uh[NCLS][RDIM][NCAPS];   // 163840 B (fits 320KB WGP LDS)
    __shared__ float sv[NCLS * RDIM];
    __shared__ float vsh[NCLS * RDIM];

    const int b = blockIdx.x;
    const int t = threadIdx.x;

    float u[2][PDIM];
#pragma unroll
    for (int j = 0; j < 2; ++j) {
        const int n = t + j * 256;
        const int cap = n >> 4, p = n & 15;
        float sq = 0.f;
#pragma unroll
        for (int d = 0; d < PDIM; ++d) {
            float x = G[(size_t)(b * 16 + p) * 256 + cap * 8 + d];
            u[j][d] = x; sq += x * x;
        }
        float scale = (sq / (1.f + sq)) * rsqrtf(sq + 1e-8f);
#pragma unroll
        for (int d = 0; d < PDIM; ++d) u[j][d] *= scale;

        const float* wn_ = Wr + (size_t)n * NCLS * RDIM * PDIM;
#pragma unroll
        for (int o = 0; o < NCLS; ++o)
            for (int dd = 0; dd < RDIM; ++dd) {
                const float* w = wn_ + (o * RDIM + dd) * PDIM;
                float a = 0.f;
#pragma unroll
                for (int din = 0; din < PDIM; ++din) a += w[din] * u[j][din];
                uh[o][dd][n] = a;   // [o][d][n]: lane-contiguous -> conflict-free
            }
    }
    float blog[2][NCLS] = {};
    __syncthreads();

    for (int it = 0; it < 3; ++it) {
        float c[2][NCLS];
#pragma unroll
        for (int j = 0; j < 2; ++j) {
            float mx = blog[j][0];
            for (int o = 1; o < NCLS; ++o) mx = fmaxf(mx, blog[j][o]);
            float s = 0.f;
            for (int o = 0; o < NCLS; ++o) { c[j][o] = __expf(blog[j][o] - mx); s += c[j][o]; }
            float inv = 1.f / s;
            for (int o = 0; o < NCLS; ++o) c[j][o] *= inv;
        }
        if (t < NCLS * RDIM) sv[t] = 0.f;
        __syncthreads();
        const int n0 = t, n1 = t + 256;
#pragma unroll
        for (int o = 0; o < NCLS; ++o)
            for (int dd = 0; dd < RDIM; ++dd) {
                float ps = c[0][o] * uh[o][dd][n0] + c[1][o] * uh[o][dd][n1];
                atomicAdd(&sv[o * RDIM + dd], ps);   // ds_add_f32
            }
        __syncthreads();
        if (t < NCLS) {
            float sq = 0.f;
            for (int dd = 0; dd < RDIM; ++dd) { float x = sv[t * RDIM + dd]; sq += x * x; }
            float scale = (sq / (1.f + sq)) * rsqrtf(sq + 1e-8f);
            for (int dd = 0; dd < RDIM; ++dd) vsh[t * RDIM + dd] = sv[t * RDIM + dd] * scale;
        }
        __syncthreads();
        if (it < 2) {
#pragma unroll
            for (int j = 0; j < 2; ++j) {
                const int n = t + j * 256;
                for (int o = 0; o < NCLS; ++o) {
                    float agr = 0.f;
                    for (int dd = 0; dd < RDIM; ++dd) agr += uh[o][dd][n] * vsh[o * RDIM + dd];
                    blog[j][o] += agr;
                }
            }
            __syncthreads();
        }
    }

    if (t < NCLS) {
        float sq = 0.f;
        for (int dd = 0; dd < RDIM; ++dd) { float x = vsh[t * RDIM + dd]; sq += x * x; }
        out_probs[(size_t)b * NCLS + t] = sqrtf(sq);
    }
    if (t < NCLS * RDIM)
        out_v[(size_t)b * NCLS * RDIM + t] = vsh[t];
}

// ---------------- launcher ----------------
extern "C" void kernel_launch(void* const* d_in, const int* in_sizes, int n_in,
                              void* d_out, int out_size, void* d_ws, size_t ws_size,
                              hipStream_t stream) {
    const float* x     = (const float*)d_in[0];  // [1024][768]
    const float* W1    = (const float*)d_in[1];  // [768][9216]
    const float* b1    = (const float*)d_in[2];  // [9216]
    const float* convW = (const float*)d_in[3];  // [256][2304]  (OIHW flattened)
    const float* convB = (const float*)d_in[4];  // [256]
    const float* Wr    = (const float*)d_in[5];  // [512][5][16][8]
    float* out = (float*)d_out;                  // 1024*5 probs ++ 1024*5*16 v

    char* ws = (char*)d_ws;
    auto carve = [&](size_t bytes) -> char* {
        char* p = ws;
        ws += (bytes + 255) & ~(size_t)255;
        return p;
    };
    bf16_t* xb  = (bf16_t*)carve((size_t)B_SZ * IN_DIM * 2);     // 1.5 MB
    bf16_t* W1T = (bf16_t*)carve((size_t)PROJ * IN_DIM * 2);     // 14.2 MB
    bf16_t* hbf = (bf16_t*)carve((size_t)B_SZ * PROJ * 2);       // 18.9 MB
    bf16_t* cwb = (bf16_t*)carve((size_t)CONV_N * CONV_K * 2);   // 1.2 MB
    float*  G   = (float*)carve((size_t)CONV_M * CONV_N * 4);    // 16.8 MB
    (void)in_sizes; (void)n_in; (void)out_size; (void)ws_size;

    // precision conversion passes
    {
        int n4 = B_SZ * IN_DIM / 4;
        cvt4_f32_bf16<<<(n4 + 255) / 256, 256, 0, stream>>>(x, xb, n4);
    }
    transpose_cvt<<<dim3(PROJ / 32, IN_DIM / 32), 256, 0, stream>>>(W1, W1T, IN_DIM, PROJ);
    {
        int n4 = CONV_N * CONV_K / 4;
        cvt4_f32_bf16<<<(n4 + 255) / 256, 256, 0, stream>>>(convW, cwb, n4);
    }

    // GEMM1: h = relu(x @ W1 + b1) -> bf16 [1024][9216]
    gemm_wmma<true, true, false><<<dim3(PROJ / 128, B_SZ / 128), 256, 0, stream>>>(
        xb, nullptr, W1T, b1, nullptr, hbf, B_SZ, PROJ, IN_DIM);

    // GEMM2: conv as implicit-im2col GEMM -> f32 G[16384][256] (+convB)
    gemm_wmma<false, false, true><<<dim3(CONV_N / 128, CONV_M / 128), 256, 0, stream>>>(
        nullptr, hbf, cwb, convB, G, nullptr, CONV_M, CONV_N, CONV_K);

    // fused squash + u_hat + 3-iter dynamic routing, u_hat resident in LDS
    routing_kernel<<<B_SZ, 256, 0, stream>>>(G, Wr, out, out + (size_t)B_SZ * NCLS);
}
